// RTRBM_841813590523
// MI455X (gfx1250) — compile-verified, matrix operand order checked by reading, not run on hardware
//
#include <hip/hip_runtime.h>

// ---------------------------------------------------------------------------
// Types for CDNA5 WMMA / TDM builtins
// ---------------------------------------------------------------------------
typedef __bf16 v16bf __attribute__((ext_vector_type(16)));
typedef __bf16 v8bf  __attribute__((ext_vector_type(8)));
typedef float  v8f   __attribute__((ext_vector_type(8)));
typedef unsigned int v4u __attribute__((ext_vector_type(4)));
typedef int v8i __attribute__((ext_vector_type(8)));
typedef int v4i __attribute__((ext_vector_type(4)));

#define V_DIM 5000
#define T_DIM 10000           // multiple of 16 -> every 16-row WMMA tile fully in/out
#define H_DIM 1024
#define K_PAD 5120            // V rounded up to multiple of 32 (zero padded)
#define M_TILE 128
#define T_PAD (79 * M_TILE)   // 10112 >= 10000
#define SCAN_WGS 16
#define SCAN_ROWS (H_DIM / SCAN_WGS)   // 64 rows of U per workgroup
#define U_LDS_STRIDE 1028              // dwords; 1024+4 pad breaks LDS bank conflicts

__device__ __forceinline__ unsigned short f2bf(float f) {
    union { float f; unsigned u; } v; v.f = f;
    unsigned r = v.u + 0x7FFFu + ((v.u >> 16) & 1u);   // round-to-nearest-even
    return (unsigned short)(r >> 16);
}

__device__ __forceinline__ float fast_sigmoid(float x) {
    return 1.0f / (1.0f + __expf(-x));
}

// ---------------------------------------------------------------------------
// Tiled transpose + fp32 -> bf16 convert, zero padding.
//   dst[r][c] = (c < srows && r < scols) ? src[c][r] : 0
// ---------------------------------------------------------------------------
__global__ __launch_bounds__(256) void transpose_bf16_kernel(
    const float* __restrict__ src, unsigned short* __restrict__ dst,
    int srows, int scols, int drows, int dcols) {
    __shared__ unsigned short tile[32][33];
    const int c0 = blockIdx.x * 32;   // dst col base == src row base
    const int r0 = blockIdx.y * 32;   // dst row base == src col base
    const int tx = threadIdx.x;       // 0..31
    const int ty = threadIdx.y;       // 0..7
#pragma unroll
    for (int i = ty; i < 32; i += 8) {
        const int sr = c0 + i, sc = r0 + tx;
        float f = (sr < srows && sc < scols) ? src[(size_t)sr * scols + sc] : 0.0f;
        tile[i][tx] = f2bf(f);
    }
    __syncthreads();
#pragma unroll
    for (int i = ty; i < 32; i += 8) {
        const int dr = r0 + i, dc = c0 + tx;
        if (dr < drows && dc < dcols)
            dst[(size_t)dr * dcols + dc] = tile[tx][i];
    }
}

// ---------------------------------------------------------------------------
// bf16 WMMA GEMM:  C[T x H] = A[Tpad x Kpad] * Bt[H x Kpad]^T
// 256 threads = 8 waves; block tile 128x128; wave tile 32x64 (2x4 WMMAs).
// Per-lane fragment load follows the 16-bit A-matrix 16x32 layout:
//   lanes 0-15 : row m = lane,    K = {0..7, 16..23}
//   lanes 16-31: row m = lane-16, K = {8..15, 24..31}
// All strides compile-time; fragment offsets are instruction immediates.
// ---------------------------------------------------------------------------
__global__ __launch_bounds__(256) void gemm_bf16_wmma(
    const unsigned short* __restrict__ A, const unsigned short* __restrict__ Bt,
    float* __restrict__ C) {
    const int lane  = threadIdx.x & 31;
    const int wave  = threadIdx.x >> 5;        // 0..7
    const int wm    = wave & 3;                // 4 waves along M
    const int wn    = wave >> 2;               // 2 waves along N
    const int mBase = blockIdx.x * 128 + wm * 32;
    const int nBase = blockIdx.y * 128 + wn * 64;
    const int row16 = lane & 15;
    const int koff  = (lane & 16) ? 8 : 0;

    const __bf16* pa = reinterpret_cast<const __bf16*>(A)
                     + (size_t)(mBase + row16) * K_PAD + koff;
    const __bf16* pb = reinterpret_cast<const __bf16*>(Bt)
                     + (size_t)(nBase + row16) * K_PAD + koff;

    v8f acc[2][4] = {};
    for (int k0 = 0; k0 < K_PAD; k0 += 32) {
        v16bf a[2], b[4];
#pragma unroll
        for (int i = 0; i < 2; ++i) {
            const __bf16* p = pa + (size_t)i * 16 * K_PAD;   // imm offset 163840B
            v8bf lo = *reinterpret_cast<const v8bf*>(p);
            v8bf hi = *reinterpret_cast<const v8bf*>(p + 16);
            a[i] = __builtin_shufflevector(lo, hi, 0,1,2,3,4,5,6,7,8,9,10,11,12,13,14,15);
        }
#pragma unroll
        for (int j = 0; j < 4; ++j) {
            const __bf16* p = pb + (size_t)j * 16 * K_PAD;   // imm offsets up to 491520B
            v8bf lo = *reinterpret_cast<const v8bf*>(p);
            v8bf hi = *reinterpret_cast<const v8bf*>(p + 16);
            b[j] = __builtin_shufflevector(lo, hi, 0,1,2,3,4,5,6,7,8,9,10,11,12,13,14,15);
        }
#pragma unroll
        for (int i = 0; i < 2; ++i)
#pragma unroll
            for (int j = 0; j < 4; ++j)
                acc[i][j] = __builtin_amdgcn_wmma_f32_16x16x32_bf16(
                    false, a[i], false, b[j], (short)0, acc[i][j], false, false);
        pa += 32;
        pb += 32;
    }

    // C/D layout: VGPR r, lanes 0-15 -> (M=r, N=lane); lanes 16-31 -> (M=r+8, N=lane-16).
    // T_DIM is a multiple of 16, so each 16-row tile is fully in range or fully out:
    // one wave-uniform scalar branch, then 8 unguarded stores w/ immediate offsets.
    const int mHalf = (lane & 16) ? 8 : 0;
    const int col16 = lane & 15;
#pragma unroll
    for (int i = 0; i < 2; ++i) {
        if (mBase + i * 16 >= T_DIM) continue;               // scalar (wave-uniform)
        const int m0 = mBase + i * 16 + mHalf;
#pragma unroll
        for (int j = 0; j < 4; ++j) {
            float* p = C + (size_t)m0 * H_DIM + nBase + j * 16 + col16;
#pragma unroll
            for (int r = 0; r < 8; ++r)
                p[(size_t)r * H_DIM] = acc[i][j][r];         // imm offset r*4096B
        }
    }
}

// ---------------------------------------------------------------------------
// Persistent recurrent scan. Each of 16 workgroups owns 64 rows of U, staged
// into LDS once via the Tensor Data Mover (iterated 1x1024 tiles, padded LDS
// row stride). Per step: 4 lanes per row dot r_{t-1} (float4 LDS reads),
// shfl-reduce, sigmoid, in-place update of r over proj, device-wide barrier.
// ---------------------------------------------------------------------------
__device__ __forceinline__ void grid_sync(unsigned int* counter, unsigned int phase,
                                          unsigned int nwg) {
    __syncthreads();
    if (threadIdx.x == 0) {
        __hip_atomic_fetch_add(counter, 1u, __ATOMIC_ACQ_REL, __HIP_MEMORY_SCOPE_AGENT);
        const unsigned int target = phase * nwg;
        while (__hip_atomic_load(counter, __ATOMIC_ACQUIRE, __HIP_MEMORY_SCOPE_AGENT) < target)
            __builtin_amdgcn_s_sleep(2);
    }
    __syncthreads();
}

extern __shared__ float u_lds[];   // SCAN_ROWS * U_LDS_STRIDE floats (257 KB)

__global__ __launch_bounds__(256) void rtrbm_scan_kernel(
    const float* __restrict__ U, const float* __restrict__ b_h,
    const float* __restrict__ b_init, float* __restrict__ r,   // r pre-filled with proj
    unsigned int* __restrict__ counter, int T, int nwg) {
    const int rBase = blockIdx.x * SCAN_ROWS;

    // ---- Stage this WG's 64 U rows into LDS ------------------------------
#if __has_builtin(__builtin_amdgcn_tensor_load_to_lds)
    if (threadIdx.x < 32) {   // one wave issues the TDM descriptor (SGPR-uniform)
        const unsigned long long ga =
            (unsigned long long)(const void*)(U + (size_t)rBase * H_DIM);
        const unsigned ldsBase = __builtin_amdgcn_groupstaticsize(); // dyn-LDS start
        v4u g0;
        g0[0] = 1u;                                          // count=1, user descriptor
        g0[1] = ldsBase;                                     // lds_addr (bytes)
        g0[2] = (unsigned)(ga & 0xFFFFFFFFu);                // global_addr[31:0]
        g0[3] = (unsigned)((ga >> 32) & 0x01FFFFFFu) | (2u << 30);  // addr[56:32]|type=2
        v8i g1;
        g1[0] = (2 << 16) | (1 << 19);     // data_size=4B, iterate_enable=1
        g1[1] = (int)(1024u << 16);        // tensor_dim0 = 1024 (bits 79:48)
        g1[2] = (int)(1024u << 16);        // tensor_dim1 = 1024 (bits 111:80)
        g1[3] = (int)(1024u << 16);        // tile_dim0   = 1024 (bits 127:112)
        g1[4] = 1;                         // tile_dim1=1, tile_dim2=0
        g1[5] = 1024;                      // tensor_dim0_stride
        g1[6] = 0; g1[7] = 0;
        v4i g2;
        g2[0] = 0;                         // tensor_dim2
        g2[1] = U_LDS_STRIDE;              // lds_addr_increment (data_size units)
        g2[2] = 1024;                      // global_addr_increment (one row)
        g2[3] = (int)((unsigned)(SCAN_ROWS - 1) << 16);   // iterate_count = 63 -> 64x
        v4i g3 = {0, 0, 0, 0};
#if defined(__clang_major__) && __clang_major__ >= 23
        v8i gx = {0, 0, 0, 0, 0, 0, 0, 0};
        __builtin_amdgcn_tensor_load_to_lds(g0, g1, g2, g3, gx, 0);
#else
        __builtin_amdgcn_tensor_load_to_lds(g0, g1, g2, g3, 0);
#endif
        __builtin_amdgcn_s_wait_tensorcnt(0);
    }
#else
    for (int e = threadIdx.x; e < SCAN_ROWS * H_DIM; e += blockDim.x)
        u_lds[(e / H_DIM) * U_LDS_STRIDE + (e % H_DIM)] =
            U[(size_t)(rBase + e / H_DIM) * H_DIM + (e % H_DIM)];
#endif
    __syncthreads();

    // ---- t = 0 : r0 = sigmoid(proj0 + b_init) ----------------------------
    for (int j = (int)threadIdx.x; j < SCAN_ROWS; j += (int)blockDim.x) {
        const int col = rBase + j;
        r[col] = fast_sigmoid(r[col] + b_init[col]);
    }
    __threadfence();
    grid_sync(counter, 1u, (unsigned)nwg);

    // ---- t = 1..T-1 ------------------------------------------------------
    const int sub  = threadIdx.x & 3;     // 4 lanes per row
    const int rloc = threadIdx.x >> 2;    // 0..63
    const int col  = rBase + rloc;
    const float bh = b_h[col];
    const float4* up = reinterpret_cast<const float4*>(
        u_lds + (size_t)rloc * U_LDS_STRIDE + sub * 256);

    for (int t = 1; t < T; ++t) {
        const float4* rp = reinterpret_cast<const float4*>(
            r + (size_t)(t - 1) * H_DIM + sub * 256);
        float sum = 0.0f;
#pragma unroll 8
        for (int i = 0; i < 64; ++i) {
            const float4 u = up[i];
            const float4 x = rp[i];
            sum += u.x * x.x + u.y * x.y + u.z * x.z + u.w * x.w;
        }
        sum += __shfl_xor(sum, 1, 32);
        sum += __shfl_xor(sum, 2, 32);
        if (sub == 0) {
            float* out = r + (size_t)t * H_DIM + col;
            *out = fast_sigmoid(*out + bh + sum);   // *out holds proj[t][col]
        }
        __threadfence();
        grid_sync(counter, (unsigned)(t + 1), (unsigned)nwg);
    }
}

// ---------------------------------------------------------------------------
// Launch
// ---------------------------------------------------------------------------
extern "C" void kernel_launch(void* const* d_in, const int* in_sizes, int n_in,
                              void* d_out, int out_size, void* d_ws, size_t ws_size,
                              hipStream_t stream) {
    (void)in_sizes; (void)n_in; (void)out_size; (void)ws_size;
    const float* v      = (const float*)d_in[0];  // V x T
    const float* W      = (const float*)d_in[1];  // V x H
    const float* U      = (const float*)d_in[2];  // H x H
    const float* b_h    = (const float*)d_in[4];  // H
    const float* b_init = (const float*)d_in[5];  // H
    float* r = (float*)d_out;                     // T x H (proj, then r in place)

    char* ws = (char*)d_ws;
    unsigned int* counter = (unsigned int*)ws;
    unsigned short* Abf  = (unsigned short*)(ws + 256);
    const size_t Abytes  = (size_t)T_PAD * K_PAD * sizeof(unsigned short);
    unsigned short* Btbf = (unsigned short*)(ws + 256 + Abytes);

    hipMemsetAsync(counter, 0, 256, stream);

    const dim3 tb(32, 8);
    transpose_bf16_kernel<<<dim3(K_PAD / 32, T_PAD / 32), tb, 0, stream>>>(
        v, Abf, V_DIM, T_DIM, T_PAD, K_PAD);
    transpose_bf16_kernel<<<dim3(K_PAD / 32, H_DIM / 32), tb, 0, stream>>>(
        W, Btbf, V_DIM, H_DIM, H_DIM, K_PAD);

    gemm_bf16_wmma<<<dim3(T_PAD / M_TILE, H_DIM / M_TILE), 256, 0, stream>>>(
        Abf, Btbf, r);

    const size_t ldsBytes = (size_t)SCAN_ROWS * U_LDS_STRIDE * sizeof(float);
    rtrbm_scan_kernel<<<SCAN_WGS, 256, ldsBytes, stream>>>(
        U, b_h, b_init, r, counter, T_DIM, SCAN_WGS);
}
